// SimTable_37383395344623
// MI455X (gfx1250) — matrix operand order repaired and checked by reading, not compiled, not run on hardware
//
#include <hip/hip_runtime.h>
#include <hip/hip_bf16.h>
#include <math.h>

// ---------------------------------------------------------------------------
// Problem constants (from reference: B=16, N=2048, D=768, fp32)
// ---------------------------------------------------------------------------
constexpr int Bc = 16;
constexpr int Nc = 2048;
constexpr int Dc = 768;
constexpr float S_SCALE = 40.0f;

typedef __attribute__((ext_vector_type(16))) __bf16 v16bf;
typedef __attribute__((ext_vector_type(8)))  __bf16 v8bf;
typedef __attribute__((ext_vector_type(4)))  __bf16 v4bf;
typedef __attribute__((ext_vector_type(8)))  float  v8f;

__device__ __forceinline__ float sigmoidf_(float x) {
    return 1.0f / (1.0f + __expf(-x));
}

// ---------------------------------------------------------------------------
// Kernel 1: L2-normalize each row (length D) and emit bf16.
// One wave32 per row. Lane handles 6 float4s strided by 32 lanes (coalesced).
// ---------------------------------------------------------------------------
__global__ void normalize_to_bf16_kernel(const float* __restrict__ x,
                                         __bf16* __restrict__ nx,
                                         int nrows) {
    const int wid  = (blockIdx.x * blockDim.x + threadIdx.x) >> 5;
    const int lane = threadIdx.x & 31;
    if (wid >= nrows) return;

    const float4* row = reinterpret_cast<const float4*>(x + (size_t)wid * Dc);
    float4 v[6];
    float ss = 0.0f;
#pragma unroll
    for (int i = 0; i < 6; ++i) {
        v[i] = row[lane + 32 * i];
        ss += v[i].x * v[i].x + v[i].y * v[i].y + v[i].z * v[i].z + v[i].w * v[i].w;
    }
    // wave32 butterfly reduction
#pragma unroll
    for (int m = 16; m >= 1; m >>= 1) ss += __shfl_xor(ss, m, 32);
    const float s = rsqrtf(ss);  // reference has no eps

    __bf16* orow = nx + (size_t)wid * Dc;
#pragma unroll
    for (int i = 0; i < 6; ++i) {
        v4bf o;
        o[0] = (__bf16)(v[i].x * s);
        o[1] = (__bf16)(v[i].y * s);
        o[2] = (__bf16)(v[i].z * s);
        o[3] = (__bf16)(v[i].w * s);
        *reinterpret_cast<v4bf*>(orow + (size_t)(lane + 32 * i) * 4) = o;
    }
}

// ---------------------------------------------------------------------------
// Kernel 2: per-wave 16-query row block, sweep all N cached columns in
// 64-column tiles (4 accumulators) with v_wmma_f32_16x16x32_bf16.
// The k-loop is rolled, software-pipelined, and unrolled by TWO pipeline
// stages so the fragment double-buffers ping-pong by register assignment
// (no v_mov rotation, no scratch spills).
//
// A fragment (16x32 bf16): lanes 0-15 row M=lane hold K 0..15 contiguous,
//   lanes 16-31 row M=lane-16 hold K 16..31 contiguous.
// B fragment (32x16 bf16): lanes 0-15 col N=lane hold K 0..15 contiguous,
//   lanes 16-31 col N=lane-16 hold K 16..31 contiguous.  nc is row-major
//   [n][k] so columns are already K-contiguous: plain b128 loads, no transpose.
// ---------------------------------------------------------------------------
#define NT 4   // column tiles per n-iteration (64 columns)

__device__ __forceinline__ v16bf load_a_frag(const __bf16* arow, int k) {
    v8bf alo = *reinterpret_cast<const v8bf*>(arow + k);
    v8bf ahi = *reinterpret_cast<const v8bf*>(arow + k + 8);
    return __builtin_shufflevector(alo, ahi,
        0, 1, 2, 3, 4, 5, 6, 7, 8, 9, 10, 11, 12, 13, 14, 15);
}

struct Frag {
    v16bf a;
    v16bf b[NT];
};

__device__ __forceinline__ Frag load_frags(const __bf16* arow,
                                           const __bf16* bcol, int k) {
    Frag f;
    f.a = load_a_frag(arow, k);
#pragma unroll
    for (int t = 0; t < NT; ++t)
        f.b[t] = *reinterpret_cast<const v16bf*>(bcol + (size_t)t * 16 * Dc + k);
    return f;
}

__device__ __forceinline__ void wmma_step(v8f acc[NT], const Frag& f) {
#pragma unroll
    for (int t = 0; t < NT; ++t)
        acc[t] = __builtin_amdgcn_wmma_f32_16x16x32_bf16(
            false, f.a, false, f.b[t], (short)0, acc[t], false, false);
}

__global__ void simtop_kernel(const __bf16* __restrict__ nq,
                              const __bf16* __restrict__ nc,
                              const float* __restrict__ thr,
                              float* __restrict__ reuse_map,
                              int* __restrict__ best_idx) {
    const int wid    = (blockIdx.x * blockDim.x + threadIdx.x) >> 5;
    const int lane   = threadIdx.x & 31;
    const int laneLo = lane & 15;
    const int half   = lane >> 4;

    const int blocksPerBatch = Nc / 16;
    const int b  = wid / blocksPerBatch;
    const int r0 = (wid % blocksPerBatch) * 16;
    if (b >= Bc) return;

    // A: this lane's row pointer, pre-offset by the half's K chunk start.
    const __bf16* arow = nq + ((size_t)(b * Nc + r0 + laneLo)) * Dc + half * 16;
    const __bf16* Bb   = nc + (size_t)b * Nc * Dc;

    float bestVal[8];
    int   bestIdxL[8];
#pragma unroll
    for (int r = 0; r < 8; ++r) { bestVal[r] = -INFINITY; bestIdxL[r] = 0; }

#pragma unroll 1
    for (int n0 = 0; n0 < Nc; n0 += 16 * NT) {
        // B column pointer for this lane: column n0+laneLo, K chunk half*16.
        const __bf16* bcol = Bb + ((size_t)(n0 + laneLo)) * Dc + half * 16;
        if (n0 + 16 * NT < Nc)
            __builtin_prefetch(
                (const void*)(Bb + ((size_t)(n0 + 16 * NT + laneLo)) * Dc), 0, 0);

        v8f acc[NT];
#pragma unroll
        for (int t = 0; t < NT; ++t) acc[t] = (v8f){};

        // ---- two-stage software-pipelined K loop (24 steps of 32) ----
        Frag f0 = load_frags(arow, bcol, 0);
#pragma unroll 1
        for (int k0 = 0; k0 < Dc - 64; k0 += 64) {
            Frag f1 = load_frags(arow, bcol, k0 + 32);  // stage 1 loads
            wmma_step(acc, f0);                         // step k0
            f0 = load_frags(arow, bcol, k0 + 64);       // stage 0 loads
            wmma_step(acc, f1);                         // step k0+32
        }
        {   // epilogue: steps Dc-64 (already in f0) and Dc-32
            Frag f1 = load_frags(arow, bcol, Dc - 32);
            wmma_step(acc, f0);
            wmma_step(acc, f1);
        }

        // ---- running per-lane top-1 ----
#pragma unroll
        for (int t = 0; t < NT; ++t) {
            const int col = n0 + t * 16 + laneLo;  // this lane's column in C
#pragma unroll
            for (int r = 0; r < 8; ++r) {
                const float cv = acc[t][r];
                if (cv > bestVal[r]) { bestVal[r] = cv; bestIdxL[r] = col; }
            }
        }
    }

    // Reduce across the 16 lanes of each half (xor 1,2,4,8 stays in-half).
#pragma unroll
    for (int m = 1; m <= 8; m <<= 1) {
#pragma unroll
        for (int r = 0; r < 8; ++r) {
            const float ov = __shfl_xor(bestVal[r], m, 32);
            const int   oi = __shfl_xor(bestIdxL[r], m, 32);
            if (ov > bestVal[r]) { bestVal[r] = ov; bestIdxL[r] = oi; }
        }
    }

    // Lane 0 writes rows r0..r0+7, lane 16 writes rows r0+8..r0+15.
    if (laneLo == 0) {
#pragma unroll
        for (int r = 0; r < 8; ++r) {
            const int row = r0 + half * 8 + r;       // position within N
            const int g   = b * Nc + row;            // flat [B,N] index
            const float t   = sigmoidf_(thr[row]);   // learned threshold
            const float map = sigmoidf_(S_SCALE * (bestVal[r] - t));
            reuse_map[g] = map;
            best_idx[g]  = bestIdxL[r];
        }
    }
}

// ---------------------------------------------------------------------------
// Kernel 3: blend. Pure bandwidth: wave per row, float4 streams.
// ---------------------------------------------------------------------------
__global__ void blend_kernel(const float* __restrict__ q_in,
                             const float* __restrict__ q_val,
                             const float* __restrict__ c_in,
                             const float* __restrict__ c_val,
                             const float* __restrict__ reuse_map,
                             const int* __restrict__ best_idx,
                             float* __restrict__ out_in,
                             float* __restrict__ out_val) {
    const int wid  = (blockIdx.x * blockDim.x + threadIdx.x) >> 5;
    const int lane = threadIdx.x & 31;
    if (wid >= Bc * Nc) return;

    const int b = wid / Nc;
    const float rm = reuse_map[wid];
    const float qm = 1.0f - rm;
    const int j = best_idx[wid];

    const float4* qi = reinterpret_cast<const float4*>(q_in  + (size_t)wid * Dc);
    const float4* qv = reinterpret_cast<const float4*>(q_val + (size_t)wid * Dc);
    const float4* ci = reinterpret_cast<const float4*>(c_in  + ((size_t)(b * Nc + j)) * Dc);
    const float4* cv = reinterpret_cast<const float4*>(c_val + ((size_t)(b * Nc + j)) * Dc);
    float4* oi = reinterpret_cast<float4*>(out_in  + (size_t)wid * Dc);
    float4* ov = reinterpret_cast<float4*>(out_val + (size_t)wid * Dc);

#pragma unroll
    for (int i = 0; i < 6; ++i) {
        const int p = lane + 32 * i;
        const float4 a = qi[p], bb = ci[p];
        const float4 c = qv[p], dd = cv[p];
        float4 r1, r2;
        r1.x = qm * a.x + rm * bb.x;  r1.y = qm * a.y + rm * bb.y;
        r1.z = qm * a.z + rm * bb.z;  r1.w = qm * a.w + rm * bb.w;
        r2.x = qm * c.x + rm * dd.x;  r2.y = qm * c.y + rm * dd.y;
        r2.z = qm * c.z + rm * dd.z;  r2.w = qm * c.w + rm * dd.w;
        oi[p] = r1;
        ov[p] = r2;
    }
}

// ---------------------------------------------------------------------------
// Launch
// ---------------------------------------------------------------------------
extern "C" void kernel_launch(void* const* d_in, const int* in_sizes, int n_in,
                              void* d_out, int out_size, void* d_ws, size_t ws_size,
                              hipStream_t stream) {
    const float* query_input   = (const float*)d_in[0];
    const float* query_value   = (const float*)d_in[1];
    const float* cached_input  = (const float*)d_in[2];
    const float* cached_value  = (const float*)d_in[3];
    const float* sim_threshold = (const float*)d_in[4];

    float* out = (float*)d_out;
    float* reuse_map = out;                                   // [B,N]
    float* out_in    = out + (size_t)Bc * Nc;                 // [B,N,D]
    float* out_val   = out + (size_t)Bc * Nc * (1 + Dc);      // [B,N,D]

    // workspace layout: nq bf16 | nc bf16 | argmax int
    char* ws = (char*)d_ws;
    const size_t bfBytes = (size_t)Bc * Nc * Dc * sizeof(__bf16);
    __bf16* nq = (__bf16*)ws;
    __bf16* nc = (__bf16*)(ws + bfBytes);
    int* best_idx = (int*)(ws + 2 * bfBytes);

    const int rows = Bc * Nc;            // 32768 rows
    const int wavesPerBlock = 8;         // 256 threads = 8 wave32
    const int threads = wavesPerBlock * 32;

    // 1) normalize -> bf16
    {
        const int blocks = (rows + wavesPerBlock - 1) / wavesPerBlock;
        normalize_to_bf16_kernel<<<blocks, threads, 0, stream>>>(query_input, nq, rows);
        normalize_to_bf16_kernel<<<blocks, threads, 0, stream>>>(cached_input, nc, rows);
    }

    // 2) WMMA similarity + top-1 + threshold
    {
        const int rowBlocks = (Bc * Nc) / 16;   // 2048 waves
        const int blocks = rowBlocks / wavesPerBlock;
        simtop_kernel<<<blocks, threads, 0, stream>>>(nq, nc, sim_threshold,
                                                      reuse_map, best_idx);
    }

    // 3) blend
    {
        const int blocks = (rows + wavesPerBlock - 1) / wavesPerBlock;
        blend_kernel<<<blocks, threads, 0, stream>>>(query_input, query_value,
                                                     cached_input, cached_value,
                                                     reuse_map, best_idx,
                                                     out_in, out_val);
    }
}
